// LSTM_7954279432775
// MI455X (gfx1250) — matrix operand order repaired and checked by reading, not compiled, not run on hardware
//
#include <hip/hip_runtime.h>
#include <hip/hip_bf16.h>
#include <math.h>

#define HID   512
#define BATCH 256
#define THIST 150
#define TFUT  50
#define NVEG  4
#define NWEA  8
#define NFEAT 12

typedef _Float16 v16h __attribute__((ext_vector_type(16)));
typedef _Float16 v8h  __attribute__((ext_vector_type(8)));
typedef float    v8f  __attribute__((ext_vector_type(8)));

__device__ __forceinline__ float sigm_f(float x) {
    return 1.0f / (1.0f + __expf(-x));
}
__device__ __forceinline__ float tanh_f(float x) {
    x = fminf(fmaxf(x, -15.0f), 15.0f);
    float e = __expf(2.0f * x);
    return (e - 1.0f) / (e + 1.0f);
}

// ---------------------------------------------------------------------------
// Prep: smoothed veg + weather concat -> x_hist laid out [t][b][12]
// ---------------------------------------------------------------------------
__global__ void prep_kernel(const float* __restrict__ veg,
                            const float* __restrict__ wea,
                            float* __restrict__ x_hist) {
    int tid = blockIdx.x * blockDim.x + threadIdx.x;
    if (tid >= BATCH * THIST * NFEAT) return;
    int j = tid % NFEAT;
    int t = (tid / NFEAT) % THIST;
    int b = tid / (NFEAT * THIST);
    float val;
    if (j < NVEG) {
        float raw = veg[(b * THIST + t) * NVEG + j];
        float s = 0.0f;
        for (int d = -2; d <= 2; ++d) {
            int tt = t + d;
            if (tt >= 0 && tt < THIST) {
                float v = veg[(b * THIST + tt) * NVEG + j];
                if (!isnan(v)) s += v;
            }
        }
        s *= 0.2f;
        val = isnan(raw) ? s : raw;
        if (isnan(val)) val = 0.0f;
    } else {
        val = wea[(b * THIST + t) * NWEA + (j - NVEG)];
    }
    x_hist[(t * BATCH + b) * NFEAT + j] = val;
}

// Zero state buffers
__global__ void init_state_kernel(float* c0, float* c1,
                                  _Float16* h0a, _Float16* h0b,
                                  _Float16* h1a, _Float16* h1b) {
    int i = blockIdx.x * blockDim.x + threadIdx.x;
    if (i >= BATCH * HID) return;
    c0[i] = 0.0f; c1[i] = 0.0f;
    h0a[i] = (_Float16)0.0f; h0b[i] = (_Float16)0.0f;
    h1a[i] = (_Float16)0.0f; h1b[i] = (_Float16)0.0f;
}

// pred init = smoothed veg at last hist step
__global__ void init_pred_kernel(const float* __restrict__ x_hist,
                                 float* __restrict__ pred) {
    int i = blockIdx.x * blockDim.x + threadIdx.x;
    if (i >= BATCH * NVEG) return;
    int b = i >> 2, j = i & 3;
    pred[i] = x_hist[((THIST - 1) * BATCH + b) * NFEAT + j];
}

// ---------------------------------------------------------------------------
// Pack a (2048 x 512) f32 weight matrix into per-lane WMMA-B f16 fragments.
// Dest layout: [(g*32+nt)*16+kt][lane][16 halfs]; B[k][n] = W[g*512+n][k];
// lane L: n = nt*16 + (L&15); element idx covers K = kt*32 + (L>>4)*8 + idx (idx<8), kt*32 + 16 + (L>>4)*8 + idx-8 (idx>=8).
// ---------------------------------------------------------------------------
__global__ void pack_kernel(const float* __restrict__ W, _Float16* __restrict__ pk) {
    int tid = blockIdx.x * blockDim.x + threadIdx.x;
    if (tid >= 4 * HID * HID) return;
    int idx  = tid & 15;
    int lane = (tid >> 4) & 31;
    int kt   = (tid >> 9) & 15;
    int gnt  = tid >> 13;              // 0..127 = g*32 + nt
    int g  = gnt >> 5;
    int nt = gnt & 31;
    int hi = lane >> 4;
    int ng = nt * 16 + (lane & 15);
    int K  = kt * 32 + hi * 8 + (idx & 7) + ((idx >> 3) << 4);
    pk[tid] = (_Float16)W[(g * HID + ng) * HID + K];
}

// A fragment (16x32 f16) per CDNA5 ISA layout: lane L, M=L&15, hi=L>>4:
// a[0..7] = A[M][k0+hi*8 .. +7], a[8..15] = A[M][k0+16+hi*8 .. +7]
__device__ __forceinline__ v16h load_a_frag(const _Float16* __restrict__ base,
                                            int row, int k0, int hi) {
    const _Float16* p0 = base + row * HID + k0 + hi * 8;
    v8h lo = *(const v8h*)p0;
    v8h hh = *(const v8h*)(p0 + 16);
    return __builtin_shufflevector(lo, hh, 0,1,2,3,4,5,6,7,8,9,10,11,12,13,14,15);
}

// ---------------------------------------------------------------------------
// Core WMMA accumulation h @ Whh^T (K=512) into 4 gate accumulators
// ---------------------------------------------------------------------------
__device__ __forceinline__ void mm_core(const _Float16* __restrict__ a_src,
                                        const _Float16* __restrict__ pB,
                                        int m, int nt, int lane, int nl, int hi,
                                        v8f& acc0, v8f& acc1, v8f& acc2, v8f& acc3) {
    const size_t gstr = (size_t)(32 * 16 * 32) << 4;   // per-gate stride in halfs
    for (int kt = 0; kt < 16; ++kt) {
        v16h a = load_a_frag(a_src, m * 16 + nl, kt * 32, hi);
        const _Float16* pb = pB + (((size_t)(nt * 16 + kt) * 32 + lane) << 4);
        v16h b0f = *(const v16h*)(pb);
        acc0 = __builtin_amdgcn_wmma_f32_16x16x32_f16(false, a, false, b0f, (short)0, acc0, false, false);
        v16h b1f = *(const v16h*)(pb + gstr);
        acc1 = __builtin_amdgcn_wmma_f32_16x16x32_f16(false, a, false, b1f, (short)0, acc1, false, false);
        v16h b2f = *(const v16h*)(pb + 2 * gstr);
        acc2 = __builtin_amdgcn_wmma_f32_16x16x32_f16(false, a, false, b2f, (short)0, acc2, false, false);
        v16h b3f = *(const v16h*)(pb + 3 * gstr);
        acc3 = __builtin_amdgcn_wmma_f32_16x16x32_f16(false, a, false, b3f, (short)0, acc3, false, false);
    }
}

// ---------------------------------------------------------------------------
// Small-input cell body (12 features via scalar FMAs in epilogue).
// x_sel: if null, gather x from pred/wfc (decoder mode).
// ---------------------------------------------------------------------------
__device__ __forceinline__ void cell_small_body(int bid, int lane,
        const float* __restrict__ x,        // [BATCH][12] or null
        const float* __restrict__ pred,     // decoder mode
        const float* __restrict__ wfc,      // decoder mode
        int t,
        const float* __restrict__ Wih,      // [2048][12]
        const float* __restrict__ bias,     // [2048]
        const _Float16* __restrict__ pWhh,
        const _Float16* __restrict__ h_in,
        float* __restrict__ c,
        _Float16* __restrict__ h_out) {
    int m  = bid & 15;
    int nt = bid >> 4;
    int hi = lane >> 4;
    int nl = lane & 15;
    int ng = nt * 16 + nl;

    v8f acc0 = {}, acc1 = {}, acc2 = {}, acc3 = {};
    mm_core(h_in, pWhh, m, nt, lane, nl, hi, acc0, acc1, acc2, acc3);

    float w[4][NFEAT], bb[4];
    #pragma unroll
    for (int g = 0; g < 4; ++g) {
        bb[g] = bias[g * HID + ng];
        #pragma unroll
        for (int j = 0; j < NFEAT; ++j)
            w[g][j] = Wih[(g * HID + ng) * NFEAT + j];
    }
    #pragma unroll
    for (int r = 0; r < 8; ++r) {
        int row = m * 16 + hi * 8 + r;
        float xr[NFEAT];
        if (x) {
            #pragma unroll
            for (int j = 0; j < NFEAT; ++j) xr[j] = x[row * NFEAT + j];
        } else {
            #pragma unroll
            for (int j = 0; j < NFEAT; ++j)
                xr[j] = (j < NVEG) ? pred[row * NVEG + j]
                                   : wfc[(row * TFUT + t) * NWEA + (j - NVEG)];
        }
        float gi = acc0[r] + bb[0], gf = acc1[r] + bb[1];
        float gg = acc2[r] + bb[2], go = acc3[r] + bb[3];
        #pragma unroll
        for (int j = 0; j < NFEAT; ++j) {
            gi = fmaf(xr[j], w[0][j], gi);
            gf = fmaf(xr[j], w[1][j], gf);
            gg = fmaf(xr[j], w[2][j], gg);
            go = fmaf(xr[j], w[3][j], go);
        }
        float ii = sigm_f(gi), ff = sigm_f(gf);
        float gt = tanh_f(gg), oo = sigm_f(go);
        int cidx = row * HID + ng;
        float c2 = ff * c[cidx] + ii * gt;
        float h2 = oo * tanh_f(c2);
        c[cidx] = c2;
        h_out[cidx] = (_Float16)h2;
    }
}

// ---------------------------------------------------------------------------
// Big-input cell body: x@Wih^T + h@Whh^T, both via WMMA.
// ---------------------------------------------------------------------------
__device__ __forceinline__ void cell_big_body(int bid, int lane,
        const _Float16* __restrict__ x_in,
        const _Float16* __restrict__ pWih,
        const float* __restrict__ bias,
        const _Float16* __restrict__ pWhh,
        const _Float16* __restrict__ h_in,
        float* __restrict__ c,
        _Float16* __restrict__ h_out) {
    int m  = bid & 15;
    int nt = bid >> 4;
    int hi = lane >> 4;
    int nl = lane & 15;
    int ng = nt * 16 + nl;

    v8f acc0 = {}, acc1 = {}, acc2 = {}, acc3 = {};
    mm_core(x_in, pWih, m, nt, lane, nl, hi, acc0, acc1, acc2, acc3);
    mm_core(h_in, pWhh, m, nt, lane, nl, hi, acc0, acc1, acc2, acc3);

    #pragma unroll
    for (int r = 0; r < 8; ++r) {
        int row = m * 16 + hi * 8 + r;
        float gi = acc0[r] + bias[0 * HID + ng];
        float gf = acc1[r] + bias[1 * HID + ng];
        float gg = acc2[r] + bias[2 * HID + ng];
        float go = acc3[r] + bias[3 * HID + ng];
        float ii = sigm_f(gi), ff = sigm_f(gf);
        float gt = tanh_f(gg), oo = sigm_f(go);
        int cidx = row * HID + ng;
        float c2 = ff * c[cidx] + ii * gt;
        float h2 = oo * tanh_f(c2);
        c[cidx] = c2;
        h_out[cidx] = (_Float16)h2;
    }
}

// Standalone wrappers ------------------------------------------------------
__global__ void __launch_bounds__(32)
cell_small_kernel(const float* __restrict__ x,
                  const float* __restrict__ Wih, const float* __restrict__ bias,
                  const _Float16* __restrict__ pWhh,
                  const _Float16* __restrict__ h_in,
                  float* __restrict__ c, _Float16* __restrict__ h_out) {
    cell_small_body(blockIdx.x, threadIdx.x, x, nullptr, nullptr, 0,
                    Wih, bias, pWhh, h_in, c, h_out);
}

__global__ void __launch_bounds__(32)
cell_small_dec_kernel(const float* __restrict__ pred, const float* __restrict__ wfc, int t,
                      const float* __restrict__ Wih, const float* __restrict__ bias,
                      const _Float16* __restrict__ pWhh,
                      const _Float16* __restrict__ h_in,
                      float* __restrict__ c, _Float16* __restrict__ h_out) {
    cell_small_body(blockIdx.x, threadIdx.x, nullptr, pred, wfc, t,
                    Wih, bias, pWhh, h_in, c, h_out);
}

__global__ void __launch_bounds__(32)
cell_big_kernel(const _Float16* __restrict__ x_in, const _Float16* __restrict__ pWih,
                const float* __restrict__ bias, const _Float16* __restrict__ pWhh,
                const _Float16* __restrict__ h_in,
                float* __restrict__ c, _Float16* __restrict__ h_out) {
    cell_big_body(blockIdx.x, threadIdx.x, x_in, pWih, bias, pWhh, h_in, c, h_out);
}

// Fused encoder diagonal: blocks [0,512) run layer0 step t+1,
// blocks [512,1024) run layer1 step t. Both depend only on h0(t)/h1(t-1..t).
// Branch is block-uniform so EXEC stays all-ones for WMMA.
__global__ void __launch_bounds__(32)
enc_pair_kernel(const float* __restrict__ x_next,
                const float* __restrict__ Wih0, const float* __restrict__ b0,
                const _Float16* __restrict__ pWhh0,
                const _Float16* __restrict__ h0cur, float* __restrict__ c0,
                _Float16* __restrict__ h0next,
                const _Float16* __restrict__ pWih1, const float* __restrict__ b1,
                const _Float16* __restrict__ pWhh1,
                const _Float16* __restrict__ h1cur, float* __restrict__ c1,
                _Float16* __restrict__ h1next) {
    int bid = blockIdx.x;
    if (bid < 512) {
        cell_small_body(bid, threadIdx.x, x_next, nullptr, nullptr, 0,
                        Wih0, b0, pWhh0, h0cur, c0, h0next);
    } else {
        cell_big_body(bid - 512, threadIdx.x, h0cur, pWih1, b1, pWhh1,
                      h1cur, c1, h1next);
    }
}

// Decoder head: one wave per (b,j); lanes strip-mine K, reduce via shfl_xor.
__global__ void __launch_bounds__(256)
head_kernel(const _Float16* __restrict__ h1,
            const float* __restrict__ headW, const float* __restrict__ headb,
            float* __restrict__ pred, float* __restrict__ out, int t) {
    int wave = (blockIdx.x * blockDim.x + threadIdx.x) >> 5;
    int lane = threadIdx.x & 31;
    if (wave >= BATCH * NVEG) return;
    int b = wave >> 2, j = wave & 3;
    float s = 0.0f;
    #pragma unroll 4
    for (int k = lane; k < HID; k += 32)
        s = fmaf((float)h1[b * HID + k], headW[j * HID + k], s);
    #pragma unroll
    for (int off = 16; off > 0; off >>= 1)
        s += __shfl_xor(s, off, 32);
    if (lane == 0) {
        float p = pred[wave] + s + headb[j];
        pred[wave] = p;
        out[(b * TFUT + t) * NVEG + j] = p;
    }
}

// ---------------------------------------------------------------------------
extern "C" void kernel_launch(void* const* d_in, const int* in_sizes, int n_in,
                              void* d_out, int out_size, void* d_ws, size_t ws_size,
                              hipStream_t stream) {
    (void)in_sizes; (void)n_in; (void)out_size; (void)ws_size;
    const float* veg   = (const float*)d_in[0];
    const float* weah  = (const float*)d_in[1];
    const float* wfc   = (const float*)d_in[2];
    const float* Wih0  = (const float*)d_in[3];
    const float* Whh0  = (const float*)d_in[4];
    const float* b0    = (const float*)d_in[5];
    const float* Wih1  = (const float*)d_in[6];
    const float* Whh1  = (const float*)d_in[7];
    const float* b1    = (const float*)d_in[8];
    const float* headW = (const float*)d_in[9];
    const float* headb = (const float*)d_in[10];
    float* out = (float*)d_out;

    char* ws = (char*)d_ws;
    size_t off = 0;
    auto alloc = [&](size_t bytes) -> char* {
        char* p = ws + off;
        off += (bytes + 255) & ~(size_t)255;
        return p;
    };
    float*    x_hist = (float*)alloc((size_t)THIST * BATCH * NFEAT * 4);
    _Float16* h0b[2] = {(_Float16*)alloc((size_t)BATCH * HID * 2),
                        (_Float16*)alloc((size_t)BATCH * HID * 2)};
    _Float16* h1b[2] = {(_Float16*)alloc((size_t)BATCH * HID * 2),
                        (_Float16*)alloc((size_t)BATCH * HID * 2)};
    float*    c0    = (float*)alloc((size_t)BATCH * HID * 4);
    float*    c1    = (float*)alloc((size_t)BATCH * HID * 4);
    float*    pred  = (float*)alloc((size_t)BATCH * NVEG * 4);
    _Float16* pWhh0 = (_Float16*)alloc((size_t)4 * HID * HID * 2);
    _Float16* pWih1 = (_Float16*)alloc((size_t)4 * HID * HID * 2);
    _Float16* pWhh1 = (_Float16*)alloc((size_t)4 * HID * HID * 2);

    prep_kernel<<<(BATCH * THIST * NFEAT + 255) / 256, 256, 0, stream>>>(veg, weah, x_hist);
    init_state_kernel<<<(BATCH * HID + 255) / 256, 256, 0, stream>>>(c0, c1, h0b[0], h0b[1], h1b[0], h1b[1]);
    init_pred_kernel<<<4, 256, 0, stream>>>(x_hist, pred);
    pack_kernel<<<(4 * HID * HID + 255) / 256, 256, 0, stream>>>(Whh0, pWhh0);
    pack_kernel<<<(4 * HID * HID + 255) / 256, 256, 0, stream>>>(Wih1, pWih1);
    pack_kernel<<<(4 * HID * HID + 255) / 256, 256, 0, stream>>>(Whh1, pWhh1);

    // Encoder pipeline. h0(t) lives in h0b[(t+1)&1]; h1(t) in h1b[(t+1)&1].
    cell_small_kernel<<<512, 32, 0, stream>>>(x_hist, Wih0, b0, pWhh0,
                                              h0b[0], c0, h0b[1]);            // L0(0)
    for (int t = 0; t < THIST - 1; ++t) {
        // one launch = { L1(t), L0(t+1) }
        enc_pair_kernel<<<1024, 32, 0, stream>>>(
            x_hist + (size_t)(t + 1) * BATCH * NFEAT,
            Wih0, b0, pWhh0, h0b[(t + 1) & 1], c0, h0b[t & 1],
            pWih1, b1, pWhh1, h1b[t & 1], c1, h1b[(t + 1) & 1]);
    }
    {   // L1(149)
        int t = THIST - 1;
        cell_big_kernel<<<512, 32, 0, stream>>>(h0b[(t + 1) & 1], pWih1, b1, pWhh1,
                                                h1b[t & 1], c1, h1b[(t + 1) & 1]);
    }

    // Decoder: strictly sequential cell0 -> cell1 -> head per step.
    for (int td = 0; td < TFUT; ++td) {
        int s = THIST + td;
        int in = s & 1, ob = (s + 1) & 1;
        cell_small_dec_kernel<<<512, 32, 0, stream>>>(pred, wfc, td, Wih0, b0, pWhh0,
                                                      h0b[in], c0, h0b[ob]);
        cell_big_kernel<<<512, 32, 0, stream>>>(h0b[ob], pWih1, b1, pWhh1,
                                                h1b[in], c1, h1b[ob]);
        head_kernel<<<128, 256, 0, stream>>>(h1b[ob], headW, headb, pred, out, td);
    }
}
